// VectorCollapseEngine_17463337025540
// MI455X (gfx1250) — matrix operand order repaired and checked by reading, not compiled, not run on hardware
//
#include <hip/hip_runtime.h>
#include <hip/hip_bf16.h>
#include <cstdint>

#define DIM        1024
#define BATCH      16384
#define NUM_LAYERS 6
#define MT         32          // rows per workgroup (2 WMMA M-blocks)
#define NWAVES     8
#define NTHREADS   (NWAVES * 32)
#define KSTR       516         // padded row stride (uint32 units) for packed-bf16 A tiles
#define TSTR       1032        // same stride in uint16 units
#define NTILES     (DIM / 16)  // 64 column tiles
#define KSTEPS     (DIM / 32)  // 32 k-steps
// packed weight: [ntile][kstep][lane][8 dwords]
#define WPACK_U32  (NTILES * KSTEPS * 32 * 8)   // 524288 u32 = 2 MB per matrix
#define WS_NEEDED  (2ull * WPACK_U32 * 4ull)    // 4 MB

typedef __attribute__((ext_vector_type(16))) __bf16 v16bf;
typedef __attribute__((ext_vector_type(8)))  float  v8f;

union AFrag {
    v16bf    v;
    uint32_t u[8];
    uint4    q[2];
};

__device__ __forceinline__ uint32_t pack_bf16(float lo, float hi) {
    uint32_t a = __float_as_uint(lo);
    uint32_t b = __float_as_uint(hi);
    a = (a + 0x7FFFu + ((a >> 16) & 1u)) >> 16;   // round-to-nearest-even
    b = (b + 0x7FFFu + ((b >> 16) & 1u)) >> 16;
    return (a & 0xFFFFu) | (b << 16);
}

__device__ __forceinline__ uint16_t f2bf(float f) {
    uint32_t a = __float_as_uint(f);
    a = (a + 0x7FFFu + ((a >> 16) & 1u)) >> 16;
    return (uint16_t)a;
}

__device__ __forceinline__ float fast_tanh(float x) {
    x = fminf(fmaxf(x, -15.f), 15.f);
    float e = __expf(2.f * x);
    return (e - 1.f) * __builtin_amdgcn_rcpf(e + 1.f);
}

__device__ __forceinline__ float wave_sum(float v) {
#pragma unroll
    for (int o = 16; o > 0; o >>= 1) v += __shfl_xor(v, o, 32);
    return v;
}

// ---- one-time weight re-pack: fp32 W[1024][1024] -> bf16 B-fragment order ----
__global__ __launch_bounds__(256)
void pack_weights_kernel(const float* __restrict__ W1, const float* __restrict__ W2,
                         uint32_t* __restrict__ ws) {
    const float*  W  = (blockIdx.y == 0) ? W1 : W2;
    uint32_t*     Wp = ws + (size_t)blockIdx.y * WPACK_U32;
    int idx = blockIdx.x * 256 + threadIdx.x;            // 0 .. WPACK_U32-1
    int i    = idx & 7;
    int lane = (idx >> 3) & 31;
    int ks   = (idx >> 8) & 31;
    int nt   = idx >> 13;
    int n = nt * 16 + (lane & 15);
    int k = ks * 32 + ((lane >> 4) << 4) + (i << 1);
    float2 wv = *(const float2*)(W + (size_t)n * DIM + k);
    Wp[idx] = pack_bf16(wv.x, wv.y);
}

// Two 16x16 output tiles (M blocks 0/1) at columns [nt*16, nt*16+16):
// out[m][n] = sum_k A[m][k] * W[n][k].
template<bool PACKED>
__device__ __forceinline__ void gemm_tile(const uint32_t* __restrict__ Abf,
                                          const float* __restrict__ W,
                                          const uint32_t* __restrict__ Wp,
                                          int nt, int lane, v8f& acc0, v8f& acc1) {
    const int r16 = lane & 15;
    const int khi = lane >> 4;
    const uint32_t* a0 = Abf + r16 * KSTR + khi * 8;
    const uint32_t* a1 = a0 + 16 * KSTR;
    const uint32_t* wl = Wp + (size_t)nt * (KSTEPS * 32 * 8) + lane * 8;
    const float*  wrow = W + (size_t)(nt * 16 + r16) * DIM;

#pragma unroll 2
    for (int ks = 0; ks < KSTEPS; ++ks) {
        AFrag A0, A1, B;
        const uint32_t* ak0 = a0 + ks * 16;
        A0.q[0] = *(const uint4*)(ak0);
        A0.q[1] = *(const uint4*)(ak0 + 4);
        const uint32_t* ak1 = a1 + ks * 16;
        A1.q[0] = *(const uint4*)(ak1);
        A1.q[1] = *(const uint4*)(ak1 + 4);
        if (PACKED) {
            const uint32_t* w = wl + ks * 256;
            B.q[0] = *(const uint4*)(w);
            B.q[1] = *(const uint4*)(w + 4);
        } else {
#pragma unroll
            for (int i = 0; i < 8; ++i) {
                int kk = ks * 32 + (khi << 4) + (i << 1);
                float2 wv = *(const float2*)(wrow + kk);
                B.u[i] = pack_bf16(wv.x, wv.y);
            }
        }
        acc0 = __builtin_amdgcn_wmma_f32_16x16x32_bf16(
                   false, A0.v, false, B.v, (short)0, acc0, false, false);
        acc1 = __builtin_amdgcn_wmma_f32_16x16x32_bf16(
                   false, A1.v, false, B.v, (short)0, acc1, false, false);
    }
}

template<bool PACKED>
__global__ __launch_bounds__(NTHREADS)
void collapse_kernel(const float* __restrict__ h0,
                     const float* __restrict__ W1,
                     const float* __restrict__ b1,
                     const float* __restrict__ W2,
                     const float* __restrict__ b2,
                     const float* __restrict__ anchE,
                     const float* __restrict__ anchC,
                     const float* __restrict__ anchN,
                     const uint32_t* __restrict__ Wp1,
                     const uint32_t* __restrict__ Wp2,
                     float* __restrict__ out) {
    __shared__ float    sh_h[MT * DIM];        // fp32 master state          (128 KB)
    __shared__ uint32_t sh_hbf[MT * KSTR];     // packed-bf16 h (padded)     (64.5 KB)
    __shared__ uint16_t sh_tbf[MT * TSTR];     // packed-bf16 tanh out       (64.5 KB)
    __shared__ float    sh_edir[DIM], sh_cdir[DIM], sh_ndir[DIM];
    __shared__ float    sh_s[MT], sh_ce[MT], sh_cc[MT], sh_cn[MT];
    __shared__ float    sh_red[NTHREADS];

    const int tid  = threadIdx.x;
    const int wave = tid >> 5;
    const int lane = tid & 31;
    const size_t base = (size_t)blockIdx.x * MT * DIM;

    // ---- load h0 tile ----
    for (int i = tid; i < MT * DIM; i += NTHREADS) sh_h[i] = h0[base + i];

    // ---- normalized anchor directions (block reduction) ----
    for (int a = 0; a < 3; ++a) {
        const float* anc = (a == 0) ? anchE : (a == 1) ? anchC : anchN;
        float*       dir = (a == 0) ? sh_edir : (a == 1) ? sh_cdir : sh_ndir;
        float p = 0.f;
        for (int k = tid; k < DIM; k += NTHREADS) p = fmaf(anc[k], anc[k], p);
        sh_red[tid] = p;
        __syncthreads();
        for (int s = NTHREADS / 2; s > 0; s >>= 1) {
            if (tid < s) sh_red[tid] += sh_red[tid + s];
            __syncthreads();
        }
        float invn = 1.f / fmaxf(sqrtf(sh_red[0]), 1e-12f);
        __syncthreads();
        for (int k = tid; k < DIM; k += NTHREADS) dir[k] = anc[k] * invn;
    }
    __syncthreads();

    for (int layer = 0; layer < NUM_LAYERS; ++layer) {
        // ---- step 1: per-row norm + anchor dots -> fused update coefficients ----
        for (int r = wave; r < MT; r += NWAVES) {
            const float* hr = sh_h + r * DIM;
            float s0 = 0.f, se = 0.f, sc = 0.f, sn = 0.f;
            for (int k = lane; k < DIM; k += 32) {
                float v = hr[k];
                s0 = fmaf(v, v, s0);
                se = fmaf(v, sh_edir[k], se);
                sc = fmaf(v, sh_cdir[k], sc);
                sn = fmaf(v, sh_ndir[k], sn);
            }
            s0 = wave_sum(s0); se = wave_sum(se);
            sc = wave_sum(sc); sn = wave_sum(sn);
            if (lane == 0) {
                float ih = 1.f / fmaxf(sqrtf(s0), 1e-12f);
                float ae = se * ih, ac = sc * ih, an = sn * ih;
                // ||h - dir||^2 = ||h||^2 - 2 h.dir + 1
                float fe = 1.f / fmaxf(sqrtf(fmaxf(s0 - 2.f * se + 1.f, 0.f)), 1e-12f);
                float fc = 1.f / fmaxf(sqrtf(fmaxf(s0 - 2.f * sc + 1.f, 0.f)), 1e-12f);
                float fn = 1.f / fmaxf(sqrtf(fmaxf(s0 - 2.f * sn + 1.f, 0.f)), 1e-12f);
                float ce = 0.1f * (1.f - ae) * fe;
                float cc = 0.1f * (1.f - ac) * fc;
                float bnd = fminf(fmaxf(1.f - fabsf(ae - ac), 0.f), 1.f);
                float cn = (0.05f * (1.f - an) + 0.05f * bnd) * fn;
                sh_s[r]  = 1.f - ce - cc - cn;
                sh_ce[r] = ce; sh_cc[r] = cc; sh_cn[r] = cn;
            }
        }
        __syncthreads();

        // ---- step 2: h -> packed bf16 ----
        for (int i = tid; i < MT * (DIM / 2); i += NTHREADS) {
            int r = i >> 9, c = i & 511;
            sh_hbf[r * KSTR + c] = pack_bf16(sh_h[r * DIM + 2 * c], sh_h[r * DIM + 2 * c + 1]);
        }
        __syncthreads();

        // ---- step 3: GEMM1  t = tanh(h @ W1^T + b1)  (bf16 out) ----
        for (int t = 0; t < 8; ++t) {
            int nt = wave * 8 + t;
            v8f acc0 = {0.f,0.f,0.f,0.f,0.f,0.f,0.f,0.f};
            v8f acc1 = {0.f,0.f,0.f,0.f,0.f,0.f,0.f,0.f};
            gemm_tile<PACKED>(sh_hbf, W1, Wp1, nt, lane, acc0, acc1);
            int n = nt * 16 + (lane & 15);
            int mbase = (lane >> 4) * 8;
            float bias = b1[n];
#pragma unroll
            for (int i = 0; i < 8; ++i) {
                sh_tbf[(mbase + i) * TSTR + n]      = f2bf(fast_tanh(acc0[i] + bias));
                sh_tbf[(16 + mbase + i) * TSTR + n] = f2bf(fast_tanh(acc1[i] + bias));
            }
        }
        __syncthreads();

        // ---- step 4: GEMM2 + fused collapse update (in place on sh_h) ----
        for (int t = 0; t < 8; ++t) {
            int nt = wave * 8 + t;
            v8f acc0 = {0.f,0.f,0.f,0.f,0.f,0.f,0.f,0.f};
            v8f acc1 = {0.f,0.f,0.f,0.f,0.f,0.f,0.f,0.f};
            gemm_tile<PACKED>((const uint32_t*)sh_tbf, W2, Wp2, nt, lane, acc0, acc1);
            int n = nt * 16 + (lane & 15);
            int mbase = (lane >> 4) * 8;
            float bias = b2[n];
            float ed = sh_edir[n], cd = sh_cdir[n], nd = sh_ndir[n];
#pragma unroll
            for (int i = 0; i < 8; ++i) {
                int m0 = mbase + i, m1 = 16 + mbase + i;
                float hv0 = sh_h[m0 * DIM + n];
                float hv1 = sh_h[m1 * DIM + n];
                sh_h[m0 * DIM + n] = hv0 * sh_s[m0] + acc0[i] + bias
                                   + sh_ce[m0] * ed + sh_cc[m0] * cd + sh_cn[m0] * nd;
                sh_h[m1 * DIM + n] = hv1 * sh_s[m1] + acc1[i] + bias
                                   + sh_ce[m1] * ed + sh_cc[m1] * cd + sh_cn[m1] * nd;
            }
        }
        __syncthreads();

        // ---- step 5: norm clamp factor ----
        for (int r = wave; r < MT; r += NWAVES) {
            const float* hr = sh_h + r * DIM;
            float s0 = 0.f;
            for (int k = lane; k < DIM; k += 32) { float v = hr[k]; s0 = fmaf(v, v, s0); }
            s0 = wave_sum(s0);
            if (lane == 0) {
                float hn = sqrtf(s0);
                sh_s[r] = (hn > 10.f) ? (10.f / (hn + 1e-8f)) : 1.f;
            }
        }
        __syncthreads();

        // ---- step 6: apply clamp ----
        for (int i = tid; i < MT * DIM; i += NTHREADS) sh_h[i] *= sh_s[i >> 10];
        __syncthreads();
    }

    // ---- write result ----
    for (int i = tid; i < MT * DIM; i += NTHREADS) out[base + i] = sh_h[i];
}

extern "C" void kernel_launch(void* const* d_in, const int* in_sizes, int n_in,
                              void* d_out, int out_size, void* d_ws, size_t ws_size,
                              hipStream_t stream) {
    (void)in_sizes; (void)n_in; (void)out_size;
    const float* h0    = (const float*)d_in[0];
    const float* W1    = (const float*)d_in[1];
    const float* b1    = (const float*)d_in[2];
    const float* W2    = (const float*)d_in[3];
    const float* b2    = (const float*)d_in[4];
    const float* anchE = (const float*)d_in[5];
    const float* anchC = (const float*)d_in[6];
    const float* anchN = (const float*)d_in[7];
    float* out = (float*)d_out;

    dim3 grid(BATCH / MT);
    dim3 block(NTHREADS);

    if (ws_size >= WS_NEEDED && d_ws != nullptr) {
        uint32_t* ws = (uint32_t*)d_ws;
        dim3 pgrid(WPACK_U32 / 256, 2);
        pack_weights_kernel<<<pgrid, 256, 0, stream>>>(W1, W2, ws);
        collapse_kernel<true><<<grid, block, 0, stream>>>(
            h0, W1, b1, W2, b2, anchE, anchC, anchN,
            ws, ws + WPACK_U32, out);
    } else {
        collapse_kernel<false><<<grid, block, 0, stream>>>(
            h0, W1, b1, W2, b2, anchE, anchC, anchN,
            nullptr, nullptr, out);
    }
}
